// ETC_26182120636450
// MI455X (gfx1250) — matrix operand order repaired and checked by reading, not compiled, not run on hardware
//
#include <hip/hip_runtime.h>
#include <hip/hip_bf16.h>

typedef __attribute__((ext_vector_type(16))) __bf16 v16bf;
typedef __attribute__((ext_vector_type(8)))  float  v8f;
typedef __attribute__((ext_vector_type(4)))  int    v4i;

// ---------------- problem constants ----------------
constexpr int B_ = 2, Sl = 4096, D = 768, H = 12, FF = 3072, L = 4;
constexpr int Sg = 256, S = Sl + Sg;          // 4352
constexpr int BL = 64, NL = 64, HD = 64;
constexpr int WL = 3 * BL + Sg;               // 448 : [l2l 192 | l2g 256]
constexpr int WG = Sg + Sl;                   // 4352: [g2g 256 | g2l 4096]
constexpr int R_L2L = 130, R_L2G = 17, R_G2G = 33, R_G2L = 17;
constexpr int BH = B_ * H;                    // 24
constexpr long long MZ = (long long)B_ * S;   // 8704 rows

// ---------------- GEMM tile config ----------------
constexpr int BM = 128, BN = 64, BK = 64, LDP = 72; // LDS row pad: 72 bf16 (=36 dwords, conflict-free)

// Async global->LDS staging (CDNA5 GLOBAL_LOAD_ASYNC_TO_LDS_B128), guarded so
// the host pass / older toolchains fall back to the synchronous copy.
#if defined(__AMDGCN__) && __has_builtin(__builtin_amdgcn_global_load_async_to_lds_b128)
#define USE_ASYNC 1
#else
#define USE_ASYNC 0
#endif

__device__ __forceinline__ void stage16(const __bf16* src, __bf16* lds_dst) {
#if USE_ASYNC
    __builtin_amdgcn_global_load_async_to_lds_b128(
        (__attribute__((address_space(1))) v4i*)(void*)src,
        (__attribute__((address_space(3))) v4i*)(void*)lds_dst, 0, 0);
#else
    *(uint4*)lds_dst = *(const uint4*)src;
#endif
}

__device__ __forceinline__ void stage_wait() {
#if USE_ASYNC
#if __has_builtin(__builtin_amdgcn_s_wait_asynccnt)
    __builtin_amdgcn_s_wait_asynccnt(0);
#else
    asm volatile("s_wait_asynccnt 0x0" ::: "memory");
#endif
#endif
    __syncthreads();
}

// =====================================================================
// Batched bf16 WMMA GEMM:  C = act( (A@B^T + bias) * alpha [+Cold] )
//   A: bf16 [M x K] row-major (lda), two-level batch strides sA1/sA2
//   B: bf16 [N x K] row-major (ldb)  (ALL B operands stored transposed)
//   C: optional f32 out (addC reads it first); Cb: optional bf16 out
// Double-buffered LDS pipeline fed by async global->LDS b128 copies.
// =====================================================================
__global__ __launch_bounds__(256) void k_gemm(
    const __bf16* __restrict__ A, long long sA1, long long sA2, int lda,
    const __bf16* __restrict__ Bm, long long sB1, long long sB2, int ldb,
    const float* __restrict__ bias,
    float* __restrict__ C, long long sC1, long long sC2, int ldc, int addC,
    __bf16* __restrict__ Cb, long long sCb1, long long sCb2, int ldcb,
    int M, int N, int K, float alpha, int relu, int nb2)
{
    __shared__ __bf16 lA[2][BM][LDP];
    __shared__ __bf16 lB[2][BN][LDP];
    const int tid  = threadIdx.x;
    const int lane = tid & 31;
    const int wave = tid >> 5;
    const int wm = wave & 3, wn = wave >> 2;      // 4 x 2 wave grid
    const int bz = blockIdx.z;
    const int i1 = bz / nb2, i2 = bz % nb2;
    A  += i1 * sA1 + i2 * sA2;
    Bm += i1 * sB1 + i2 * sB2;
    const int m0 = blockIdx.y * BM;
    const int n0 = blockIdx.x * BN;
    const int half = lane >> 4;
    const int l15  = lane & 15;

    v8f acc[2][2] = {};

    auto stage = [&](int buf, int k0) {
        // A tile: BM x BK, 8 chunks of 8 bf16 per row
        for (int i = tid; i < (BM * BK) / 8; i += 256) {
            int r = i >> 3, cb = i & 7;
            __bf16* dst = &lA[buf][r][cb * 8];
            int gr = m0 + r;
            if (gr < M) stage16(A + (long long)gr * lda + k0 + cb * 8, dst);
            else { uint4 z = {0u,0u,0u,0u}; *(uint4*)dst = z; }
        }
        // B tile: BN x BK (B stored N x K)
        for (int i = tid; i < (BN * BK) / 8; i += 256) {
            int r = i >> 3, cb = i & 7;
            __bf16* dst = &lB[buf][r][cb * 8];
            int gn = n0 + r;
            if (gn < N) stage16(Bm + (long long)gn * ldb + k0 + cb * 8, dst);
            else { uint4 z = {0u,0u,0u,0u}; *(uint4*)dst = z; }
        }
    };

    auto compute = [&](int buf) {
        union FR { uint4 u[2]; v16bf v; };
#pragma unroll
        for (int ks = 0; ks < BK; ks += 32) {
            FR fa[2], fb[2];
#pragma unroll
            for (int mt = 0; mt < 2; ++mt) {   // A lane<16: K {0..7,16..23}
                const __bf16* p = &lA[buf][wm * 32 + mt * 16 + l15][ks + half * 8];
                fa[mt].u[0] = *(const uint4*)p;
                fa[mt].u[1] = *(const uint4*)(p + 16);
            }
#pragma unroll
            for (int nt = 0; nt < 2; ++nt) {   // B lane<16: K {0..7,16..23}
                const __bf16* p = &lB[buf][wn * 32 + nt * 16 + l15][ks + half * 8];
                fb[nt].u[0] = *(const uint4*)p;
                fb[nt].u[1] = *(const uint4*)(p + 16);
            }
#pragma unroll
            for (int mt = 0; mt < 2; ++mt)
#pragma unroll
                for (int nt = 0; nt < 2; ++nt)
                    acc[mt][nt] = __builtin_amdgcn_wmma_f32_16x16x32_bf16(
                        false, fa[mt].v, false, fb[nt].v, (short)0, acc[mt][nt], false, false);
        }
    };

    int cur = 0;
    stage(0, 0);
    stage_wait();
    for (int k0 = 0; k0 < K; k0 += BK) {
        if (k0 + BK < K) stage(cur ^ 1, k0 + BK);
        compute(cur);
        stage_wait();
        cur ^= 1;
    }

    // ---- epilogue ----
    float*  Cp  = C  ? C  + i1 * sC1  + i2 * sC2  : nullptr;
    __bf16* Cbp = Cb ? Cb + i1 * sCb1 + i2 * sCb2 : nullptr;
    for (int mt = 0; mt < 2; ++mt)
        for (int nt = 0; nt < 2; ++nt) {
            int col = n0 + wn * 32 + nt * 16 + l15;
            if (col >= N) continue;
            float bv = bias ? bias[col] : 0.0f;
            for (int r = 0; r < 8; ++r) {
                int row = m0 + wm * 32 + mt * 16 + half * 8 + r;
                if (row >= M) continue;
                float v = (acc[mt][nt][r] + bv) * alpha;
                if (addC && Cp) v += Cp[(long long)row * ldc + col];
                if (relu) v = v > 0.0f ? v : 0.0f;
                if (Cp)  Cp [(long long)row * ldc  + col] = v;
                if (Cbp) Cbp[(long long)row * ldcb + col] = (__bf16)v;
            }
        }
}

// ---------------- elementwise / helper kernels ----------------
__global__ void k_f2b(const float* __restrict__ x, __bf16* __restrict__ y, long long n) {
    long long i = (long long)blockIdx.x * 256 + threadIdx.x;
    if (i < n) y[i] = (__bf16)x[i];
}

// batched transpose convert: x (nmat, rows, cols) f32 -> y (nmat, cols, rows) bf16
__global__ void k_f2bT(const float* __restrict__ x, __bf16* __restrict__ y,
                       int rows, int cols, long long total) {
    long long i = (long long)blockIdx.x * 256 + threadIdx.x;
    if (i >= total) return;
    int r = (int)(i % rows);
    long long t = i / rows;
    int c = (int)(t % cols);
    int m = (int)(t / cols);
    y[i] = (__bf16)x[((long long)m * rows + r) * cols + c];
}

__global__ void k_embed(const int* __restrict__ lid, const int* __restrict__ gid,
                        const float* __restrict__ emb,
                        float* __restrict__ z, __bf16* __restrict__ zb) {
    long long i = (long long)blockIdx.x * 256 + threadIdx.x;
    if (i >= (long long)B_ * S * D) return;
    int col = (int)(i % D);
    long long row = i / D;
    int b = (int)(row / S), s = (int)(row % S);
    int tok = (s < Sl) ? lid[b * Sl + s] : gid[b * Sg + (s - Sl)];
    float v = emb[(long long)tok * D + col];
    z[i] = v; zb[i] = (__bf16)v;
}

// (B,S,D) bf16 -> (B,H,S,HD) bf16
__global__ void k_split(const __bf16* __restrict__ x, __bf16* __restrict__ y) {
    long long i = (long long)blockIdx.x * 256 + threadIdx.x;
    if (i >= (long long)B_ * S * D) return;
    int d = (int)(i % HD);
    long long t = i / HD;
    int s = (int)(t % S);
    long long t2 = t / S;
    int h = (int)(t2 % H);
    int b = (int)(t2 / H);
    y[i] = x[((long long)b * S + s) * D + h * HD + d];
}

// (B,S,D) bf16 -> (B,H,HD,S) bf16  (head-split, transposed: for V)
__global__ void k_splitT(const __bf16* __restrict__ x, __bf16* __restrict__ y) {
    long long i = (long long)blockIdx.x * 256 + threadIdx.x;
    if (i >= (long long)B_ * S * D) return;
    int s = (int)(i % S);
    long long t = i / S;
    int d = (int)(t % HD);
    long long t2 = t / HD;
    int h = (int)(t2 % H);
    int b = (int)(t2 / H);
    y[i] = x[((long long)b * S + s) * D + h * HD + d];
}

// (B,H,S,HD) f32 -> (B,S,D) bf16
__global__ void k_merge(const float* __restrict__ o, __bf16* __restrict__ y) {
    long long i = (long long)blockIdx.x * 256 + threadIdx.x;
    if (i >= (long long)B_ * S * D) return;
    int col = (int)(i % D);
    long long row = i / D;
    int s = (int)(row % S), b = (int)(row / S);
    int h = col / HD, d = col % HD;
    y[i] = (__bf16)o[(((long long)b * H + h) * S + s) * HD + d];
}

// band gather: k3 (BH,NL,192,HD) from kh (BH,S,HD) and
//              v3T (BH,NL,HD,192) from vhT (BH,HD,S); zero pad at edges
__global__ void k_k3v3(const __bf16* __restrict__ kh, const __bf16* __restrict__ vht,
                       __bf16* __restrict__ k3, __bf16* __restrict__ v3t) {
    long long i = (long long)blockIdx.x * 256 + threadIdx.x;
    if (i >= (long long)BH * NL * 192 * HD) return;
    int d = (int)(i % HD);
    long long t = i / HD;
    int j = (int)(t % 192);
    long long t2 = t / 192;
    int n = (int)(t2 % NL);
    int bh = (int)(t2 / NL);
    int blk = n - 1 + j / BL;
    int p = j % BL;
    __bf16 kv = (__bf16)0.0f, vv = (__bf16)0.0f;
    if (blk >= 0 && blk < NL) {
        kv = kh [((long long)bh * S + blk * BL + p) * HD + d];
        vv = vht[((long long)bh * HD + d) * S + blk * BL + p];
    }
    k3[i] = kv;
    v3t[(((long long)bh * NL + n) * HD + d) * 192 + j] = vv;
}

// s[bh,row,coloff+c] = ok ? s + qt[bh,row,id] : -1e9   (ids broadcast over H)
__global__ void k_biasmask(float* __restrict__ s, int rowlen, int coloff, int ncols,
                           const int* __restrict__ ids, const float* __restrict__ qt,
                           int qld, int NQ, int edge, long long total) {
    long long i = (long long)blockIdx.x * 256 + threadIdx.x;
    if (i >= total) return;
    int c = (int)(i % ncols);
    long long t = i / ncols;
    int row = (int)(t % NQ);
    int bh = (int)(t / NQ);
    int b = bh / H;
    int id = ids[((long long)b * NQ + row) * ncols + c];
    bool ok = (id != 0);
    if (edge) {
        int n = row >> 6;
        if ((n == 0 && c < BL) || (n == NL - 1 && c >= 2 * BL)) ok = false;
    }
    long long off = ((long long)bh * NQ + row) * rowlen + coloff + c;
    if (ok) s[off] += qt[((long long)bh * NQ + row) * qld + id];
    else    s[off] = -1e9f;
}

// block-per-row softmax; writes bf16 probs in place over the f32 row
__global__ __launch_bounds__(256) void k_softmax(float* __restrict__ s, int rowlen, long long rstride) {
    __shared__ float red[256];
    float* row = s + (long long)blockIdx.x * rstride;
    const int tid = threadIdx.x;
    const int cnt = (rowlen + 255) / 256;
    float ev[17];
    float mx = -3.4e38f;
    for (int c = 0; c < cnt; ++c) {
        int j = c * 256 + tid;
        if (j < rowlen) { float v = row[j]; ev[c] = v; mx = fmaxf(mx, v); }
    }
    red[tid] = mx; __syncthreads();
    for (int o = 128; o > 0; o >>= 1) { if (tid < o) red[tid] = fmaxf(red[tid], red[tid + o]); __syncthreads(); }
    mx = red[0]; __syncthreads();
    float sum = 0.0f;
    for (int c = 0; c < cnt; ++c) {
        int j = c * 256 + tid;
        if (j < rowlen) { float e = __expf(ev[c] - mx); ev[c] = e; sum += e; }
    }
    red[tid] = sum; __syncthreads();
    for (int o = 128; o > 0; o >>= 1) { if (tid < o) red[tid] += red[tid + o]; __syncthreads(); }
    float inv = 1.0f / red[0];
    __syncthreads();
    __bf16* out = (__bf16*)row;
    for (int c = 0; c < cnt; ++c) {
        int j = c * 256 + tid;
        if (j < rowlen) out[j] = (__bf16)(ev[c] * inv);
    }
}

// z = LN(z + dlt) * g + b ; writes f32 z and bf16 zb.  D = 768 = 3*256
__global__ __launch_bounds__(256) void k_ln(const float* __restrict__ dlt,
                                            const float* __restrict__ g, const float* __restrict__ bt,
                                            float* __restrict__ z, __bf16* __restrict__ zb) {
    __shared__ float red[256];
    const long long r = blockIdx.x;
    const int tid = threadIdx.x;
    float vals[3];
    float sum = 0.0f;
    for (int c = 0; c < 3; ++c) {
        int j = c * 256 + tid;
        float v = z[r * D + j] + dlt[r * D + j];
        vals[c] = v; sum += v;
    }
    red[tid] = sum; __syncthreads();
    for (int o = 128; o > 0; o >>= 1) { if (tid < o) red[tid] += red[tid + o]; __syncthreads(); }
    float mean = red[0] / (float)D; __syncthreads();
    float vs = 0.0f;
    for (int c = 0; c < 3; ++c) { float d = vals[c] - mean; vs += d * d; }
    red[tid] = vs; __syncthreads();
    for (int o = 128; o > 0; o >>= 1) { if (tid < o) red[tid] += red[tid + o]; __syncthreads(); }
    float rinv = rsqrtf(red[0] / (float)D + 1e-5f);
    for (int c = 0; c < 3; ++c) {
        int j = c * 256 + tid;
        float o = (vals[c] - mean) * rinv * g[j] + bt[j];
        z[r * D + j] = o; zb[r * D + j] = (__bf16)o;
    }
}

// split z (B,S,D) -> [ (B,Sl,D) , (B,Sg,D) ]
__global__ void k_out(const float* __restrict__ z, float* __restrict__ out, long long n) {
    long long i = (long long)blockIdx.x * 256 + threadIdx.x;
    if (i >= n) return;
    const long long P1 = (long long)B_ * Sl * D;
    if (i < P1) {
        long long b = i / ((long long)Sl * D), r = i % ((long long)Sl * D);
        out[i] = z[(b * S) * D + r];
    } else {
        long long i2 = i - P1;
        long long b = i2 / ((long long)Sg * D), r = i2 % ((long long)Sg * D);
        out[i] = z[(b * S + Sl) * D + r];
    }
}

// ---------------- host side ----------------
static inline size_t aln(size_t x) { return (x + 255) & ~(size_t)255; }
static inline dim3 ew(long long n) { return dim3((unsigned)((n + 255) / 256)); }

static void gemm(hipStream_t st,
                 const __bf16* A, long long sA1, long long sA2, int lda,
                 const __bf16* Bm, long long sB1, long long sB2, int ldb,
                 const float* bias,
                 float* C, long long sC1, long long sC2, int ldc, int addC,
                 __bf16* Cb, long long sCb1, long long sCb2, int ldcb,
                 int M, int N, int K, float alpha, int relu, int nb1, int nb2) {
    dim3 g((N + BN - 1) / BN, (M + BM - 1) / BM, nb1 * nb2);
    k_gemm<<<g, 256, 0, st>>>(A, sA1, sA2, lda, Bm, sB1, sB2, ldb, bias,
                              C, sC1, sC2, ldc, addC, Cb, sCb1, sCb2, ldcb,
                              M, N, K, alpha, relu, nb2);
}

extern "C" void kernel_launch(void* const* d_in, const int* in_sizes, int n_in,
                              void* d_out, int out_size, void* d_ws, size_t ws_size,
                              hipStream_t stream) {
    typedef long long ll;
    const int* lid_p = (const int*)d_in[0];
    const int* gid_p = (const int*)d_in[1];
    const int* l2l_p = (const int*)d_in[2];
    const int* l2g_p = (const int*)d_in[3];
    const int* g2g_p = (const int*)d_in[4];
    const int* g2l_p = (const int*)d_in[5];
    const float* emb_p = (const float*)d_in[6];
    const float* Wq_p = (const float*)d_in[7];
    const float* Wk_p = (const float*)d_in[8];
    const float* Wv_p = (const float*)d_in[9];
    const float* Wo_p = (const float*)d_in[10];
    const float* bq_p = (const float*)d_in[11];
    const float* bk_p = (const float*)d_in[12];
    const float* bv_p = (const float*)d_in[13];
    const float* bo_p = (const float*)d_in[14];
    const float* ln1g_p = (const float*)d_in[15];
    const float* ln1b_p = (const float*)d_in[16];
    const float* ln2g_p = (const float*)d_in[17];
    const float* ln2b_p = (const float*)d_in[18];
    const float* W1_p = (const float*)d_in[19];
    const float* b1_p = (const float*)d_in[20];
    const float* W2_p = (const float*)d_in[21];
    const float* b2_p = (const float*)d_in[22];
    const float* rl2l_p = (const float*)d_in[23];
    const float* rl2g_p = (const float*)d_in[24];
    const float* rg2g_p = (const float*)d_in[25];
    const float* rg2l_p = (const float*)d_in[26];
    (void)in_sizes; (void)n_in;

    char* ws = (char*)d_ws;
    size_t off = 0;
    auto alloc = [&](size_t bytes) { size_t o = off; off = aln(off + bytes); return o; };

    float*  zf    = (float*) (ws + alloc(MZ * D * 4));
    __bf16* zb    = (__bf16*)(ws + alloc(MZ * D * 2));
    float*  tmpf  = (float*) (ws + alloc(MZ * D * 4));
    __bf16* projb = (__bf16*)(ws + alloc(MZ * D * 2));
    __bf16* qh    = (__bf16*)(ws + alloc((ll)BH * S * HD * 2));
    __bf16* kh    = (__bf16*)(ws + alloc((ll)BH * S * HD * 2));
    __bf16* vhT   = (__bf16*)(ws + alloc((ll)BH * HD * S * 2));
    float*  ohf   = (float*) (ws + alloc((ll)BH * S * HD * 4));
    float*  slf   = (float*) (ws + alloc((ll)BH * Sl * WL * 4));
    float*  sgf   = (float*) (ws + alloc((ll)BH * Sg * WG * 4));
    float*  qtf   = (float*) (ws + alloc((ll)BH * Sl * R_L2L * 4));
    __bf16* k3b   = (__bf16*)(ws + alloc((ll)BH * NL * 192 * HD * 2));
    __bf16* v3t   = (__bf16*)(ws + alloc((ll)BH * NL * HD * 192 * 2));
    __bf16* wqb   = (__bf16*)(ws + alloc((ll)L * D * D * 2));   // transposed (N x K)
    __bf16* wkb   = (__bf16*)(ws + alloc((ll)L * D * D * 2));
    __bf16* wvb   = (__bf16*)(ws + alloc((ll)L * D * D * 2));
    __bf16* wob   = (__bf16*)(ws + alloc((ll)L * D * D * 2));
    __bf16* w1b   = (__bf16*)(ws + alloc((ll)L * D * FF * 2));  // (FF x D) per layer
    __bf16* w2b   = (__bf16*)(ws + alloc((ll)L * FF * D * 2));  // (D x FF) per layer
    __bf16* ffhb  = (__bf16*)(ws + alloc(MZ * FF * 2));
    __bf16* rl2lb = (__bf16*)(ws + alloc((ll)L * R_L2L * HD * 2));
    __bf16* rl2gb = (__bf16*)(ws + alloc((ll)L * R_L2G * HD * 2));
    __bf16* rg2gb = (__bf16*)(ws + alloc((ll)L * R_G2G * HD * 2));
    __bf16* rg2lb = (__bf16*)(ws + alloc((ll)L * R_G2L * HD * 2));
    if (off > ws_size) return;  // workspace too small; bail safely

    // weight conversions: transpose into (N x K) bf16
    k_f2bT<<<ew((ll)L*D*D), 256, 0, stream>>>(Wq_p, wqb, D, D, (ll)L*D*D);
    k_f2bT<<<ew((ll)L*D*D), 256, 0, stream>>>(Wk_p, wkb, D, D, (ll)L*D*D);
    k_f2bT<<<ew((ll)L*D*D), 256, 0, stream>>>(Wv_p, wvb, D, D, (ll)L*D*D);
    k_f2bT<<<ew((ll)L*D*D), 256, 0, stream>>>(Wo_p, wob, D, D, (ll)L*D*D);
    k_f2bT<<<ew((ll)L*D*FF), 256, 0, stream>>>(W1_p, w1b, D, FF, (ll)L*D*FF);
    k_f2bT<<<ew((ll)L*FF*D), 256, 0, stream>>>(W2_p, w2b, FF, D, (ll)L*FF*D);
    // rel tables are already (R x K): plain convert
    k_f2b<<<ew((ll)L*R_L2L*HD), 256, 0, stream>>>(rl2l_p, rl2lb, (ll)L*R_L2L*HD);
    k_f2b<<<ew((ll)L*R_L2G*HD), 256, 0, stream>>>(rl2g_p, rl2gb, (ll)L*R_L2G*HD);
    k_f2b<<<ew((ll)L*R_G2G*HD), 256, 0, stream>>>(rg2g_p, rg2gb, (ll)L*R_G2G*HD);
    k_f2b<<<ew((ll)L*R_G2L*HD), 256, 0, stream>>>(rg2l_p, rg2lb, (ll)L*R_G2L*HD);

    k_embed<<<ew(MZ * D), 256, 0, stream>>>(lid_p, gid_p, emb_p, zf, zb);

    const ll SHD  = (ll)S * HD;      // per-(b,h) stride in qh/kh/ohf and vhT
    const ll SlHD = (ll)Sl * HD;     // offset of global tokens (row layouts)
    const int MZi = (int)MZ;

    for (int i = 0; i < L; ++i) {
        const __bf16* wq_i = wqb + (ll)i * D * D;
        const __bf16* wk_i = wkb + (ll)i * D * D;
        const __bf16* wv_i = wvb + (ll)i * D * D;
        const __bf16* wo_i = wob + (ll)i * D * D;
        const __bf16* w1_i = w1b + (ll)i * FF * D;
        const __bf16* w2_i = w2b + (ll)i * D * FF;

        // ---- projections (q scaled by 1/sqrt(HD)) ----
        gemm(stream, zb, 0, 0, D, wq_i, 0, 0, D, bq_p + i * D,
             nullptr, 0, 0, 0, 0, projb, 0, 0, D, MZi, D, D, 0.125f, 0, 1, 1);
        k_split<<<ew(MZ * D), 256, 0, stream>>>(projb, qh);
        gemm(stream, zb, 0, 0, D, wk_i, 0, 0, D, bk_p + i * D,
             nullptr, 0, 0, 0, 0, projb, 0, 0, D, MZi, D, D, 1.0f, 0, 1, 1);
        k_split<<<ew(MZ * D), 256, 0, stream>>>(projb, kh);
        gemm(stream, zb, 0, 0, D, wv_i, 0, 0, D, bv_p + i * D,
             nullptr, 0, 0, 0, 0, projb, 0, 0, D, MZi, D, D, 1.0f, 0, 1, 1);
        k_splitT<<<ew(MZ * D), 256, 0, stream>>>(projb, vhT);

        // ---- banded l2l attention ----
        k_k3v3<<<ew((ll)BH * NL * 192 * HD), 256, 0, stream>>>(kh, vhT, k3b, v3t);

        gemm(stream, qh, SHD, (ll)BL * HD, HD,                      // s_l2l
             k3b, (ll)NL * 192 * HD, (ll)192 * HD, HD, nullptr,
             slf, (ll)Sl * WL, (ll)BL * WL, WL, 0, nullptr, 0, 0, 0,
             BL, 192, HD, 1.0f, 0, BH, NL);
        gemm(stream, qh, SHD, 0, HD,                                // qt_l2l
             rl2lb + (ll)i * R_L2L * HD, 0, 0, HD, nullptr,
             qtf, (ll)Sl * R_L2L, 0, R_L2L, 0, nullptr, 0, 0, 0,
             Sl, R_L2L, HD, 1.0f, 0, BH, 1);
        k_biasmask<<<ew((ll)BH * Sl * 192), 256, 0, stream>>>(
            slf, WL, 0, 192, l2l_p, qtf, R_L2L, Sl, 1, (ll)BH * Sl * 192);

        // ---- l2g ----
        gemm(stream, qh, SHD, 0, HD, kh + SlHD, SHD, 0, HD, nullptr,
             slf + 192, (ll)Sl * WL, 0, WL, 0, nullptr, 0, 0, 0,
             Sl, Sg, HD, 1.0f, 0, BH, 1);
        gemm(stream, qh, SHD, 0, HD, rl2gb + (ll)i * R_L2G * HD, 0, 0, HD, nullptr,
             qtf, (ll)Sl * R_L2G, 0, R_L2G, 0, nullptr, 0, 0, 0,
             Sl, R_L2G, HD, 1.0f, 0, BH, 1);
        k_biasmask<<<ew((ll)BH * Sl * Sg), 256, 0, stream>>>(
            slf, WL, 192, Sg, l2g_p, qtf, R_L2G, Sl, 0, (ll)BH * Sl * Sg);

        // ---- long softmax (in place -> bf16 probs, row stride 2*WL bf16) ----
        k_softmax<<<dim3((unsigned)(BH * Sl)), 256, 0, stream>>>(slf, WL, (ll)WL);

        // ---- o_long = p_local @ v3 + p_l2g @ vg ----
        gemm(stream, (const __bf16*)slf, (ll)Sl * 2 * WL, (ll)BL * 2 * WL, 2 * WL,
             v3t, (ll)NL * HD * 192, (ll)HD * 192, 192, nullptr,
             ohf, SHD, (ll)BL * HD, HD, 0, nullptr, 0, 0, 0,
             BL, HD, 192, 1.0f, 0, BH, NL);
        gemm(stream, ((const __bf16*)slf) + 192, (ll)Sl * 2 * WL, 0, 2 * WL,
             vhT + Sl, (ll)HD * S, 0, S, nullptr,
             ohf, SHD, 0, HD, 1, nullptr, 0, 0, 0,
             Sl, HD, Sg, 1.0f, 0, BH, 1);

        // ---- global attention: [g2g | g2l] ----
        const __bf16* qg = qh + SlHD;
        gemm(stream, qg, SHD, 0, HD, kh + SlHD, SHD, 0, HD, nullptr,   // s_g2g
             sgf, (ll)Sg * WG, 0, WG, 0, nullptr, 0, 0, 0,
             Sg, Sg, HD, 1.0f, 0, BH, 1);
        gemm(stream, qg, SHD, 0, HD, rg2gb + (ll)i * R_G2G * HD, 0, 0, HD, nullptr,
             qtf, (ll)Sg * R_G2G, 0, R_G2G, 0, nullptr, 0, 0, 0,
             Sg, R_G2G, HD, 1.0f, 0, BH, 1);
        k_biasmask<<<ew((ll)BH * Sg * Sg), 256, 0, stream>>>(
            sgf, WG, 0, Sg, g2g_p, qtf, R_G2G, Sg, 0, (ll)BH * Sg * Sg);
        gemm(stream, qg, SHD, 0, HD, kh, SHD, 0, HD, nullptr,          // s_g2l
             sgf + Sg, (ll)Sg * WG, 0, WG, 0, nullptr, 0, 0, 0,
             Sg, Sl, HD, 1.0f, 0, BH, 1);
        gemm(stream, qg, SHD, 0, HD, rg2lb + (ll)i * R_G2L * HD, 0, 0, HD, nullptr,
             qtf, (ll)Sg * R_G2L, 0, R_G2L, 0, nullptr, 0, 0, 0,
             Sg, R_G2L, HD, 1.0f, 0, BH, 1);
        k_biasmask<<<ew((ll)BH * Sg * Sl), 256, 0, stream>>>(
            sgf, WG, Sg, Sl, g2l_p, qtf, R_G2L, Sg, 0, (ll)BH * Sg * Sl);

        k_softmax<<<dim3((unsigned)(BH * Sg)), 256, 0, stream>>>(sgf, WG, (ll)WG);

        gemm(stream, (const __bf16*)sgf, (ll)Sg * 2 * WG, 0, 2 * WG,      // o_glob p1
             vhT + Sl, (ll)HD * S, 0, S, nullptr,
             ohf + SlHD, SHD, 0, HD, 0, nullptr, 0, 0, 0,
             Sg, HD, Sg, 1.0f, 0, BH, 1);
        gemm(stream, ((const __bf16*)sgf) + Sg, (ll)Sg * 2 * WG, 0, 2 * WG, // o_glob p2
             vhT, (ll)HD * S, 0, S, nullptr,
             ohf + SlHD, SHD, 0, HD, 1, nullptr, 0, 0, 0,
             Sg, HD, Sl, 1.0f, 0, BH, 1);

        // ---- output projection + LN1 ----
        k_merge<<<ew(MZ * D), 256, 0, stream>>>(ohf, projb);
        gemm(stream, projb, 0, 0, D, wo_i, 0, 0, D, bo_p + i * D,
             tmpf, 0, 0, D, 0, nullptr, 0, 0, 0, MZi, D, D, 1.0f, 0, 1, 1);
        k_ln<<<dim3((unsigned)MZ), 256, 0, stream>>>(tmpf, ln1g_p + i * D, ln1b_p + i * D, zf, zb);

        // ---- FFN + LN2 ----
        gemm(stream, zb, 0, 0, D, w1_i, 0, 0, D, b1_p + i * FF,
             nullptr, 0, 0, 0, 0, ffhb, 0, 0, FF, MZi, FF, D, 1.0f, 1, 1, 1);
        gemm(stream, ffhb, 0, 0, FF, w2_i, 0, 0, FF, b2_p + i * D,
             tmpf, 0, 0, D, 0, nullptr, 0, 0, 0, MZi, D, FF, 1.0f, 0, 1, 1);
        k_ln<<<dim3((unsigned)MZ), 256, 0, stream>>>(tmpf, ln2g_p + i * D, ln2b_p + i * D, zf, zb);
    }

    k_out<<<ew((ll)out_size), 256, 0, stream>>>(zf, (float*)d_out, (ll)out_size);
}
